// T4c22GNN_73847667687964
// MI455X (gfx1250) — matrix operand order repaired and checked by reading, not compiled, not run on hardware
//
#include <hip/hip_runtime.h>

// ---------------------------------------------------------------------------
// CDNA5 (gfx1250) GNN: bf16 WMMA GEMMs (fp32 accum), pre-packed B fragments,
// BN folded into downstream weights, int32 addressing, fast-rcp swish,
// atomic scatter-mean, two-pass BatchNorm.
// ---------------------------------------------------------------------------

typedef __attribute__((ext_vector_type(16))) __bf16 v16bf;
typedef __attribute__((ext_vector_type(8)))  float  v8f;

#ifndef DEVFN
#define DEVFN __device__ __forceinline__
#endif

constexpr int Nn  = 20000;
constexpr int Ee  = 320000;
constexpr int Hh  = 64;

// ---- WMMA fragment helpers (layouts per CDNA5 ISA 7.12.2) ------------------

// A-matrix (16-bit, 16x32): element e of lane -> K index.
DEVFN int a_k_of(int e, int hi) {
  int v = e >> 1, p = e & 1;
  return 2 * v + ((v & 4) ? 8 : 0) + p + (hi ? 8 : 0);
}

// A fragment from a row-major bf16 tile (LDS or global): 8x 32-bit loads.
DEVFN v16bf load_a_tile(const __bf16* tile, int lda, int k0, int lane) {
  int r = lane & 15, hi = (lane >> 4) & 1;
  const __bf16* row = tile + r * lda + k0 + (hi ? 8 : 0);
  union { v16bf v; unsigned int u[8]; } U;
#pragma unroll
  for (int v = 0; v < 8; ++v) {
    int base = 2 * v + ((v & 4) ? 8 : 0);
    U.u[v] = *(const unsigned int*)(row + base);
  }
  return U.v;
}

// B fragments pre-packed in exact lane order: one 32-byte load per fragment.
DEVFN v16bf load_bp(const __bf16* __restrict__ P, int kt, int nt, int lane) {
  return ((const v16bf*)P)[(kt * 4 + nt) * 32 + lane];
}

DEVFN v8f wmma_bf16(v16bf a, v16bf b, v8f c) {
  return __builtin_amdgcn_wmma_f32_16x16x32_bf16(false, a, false, b, (short)0, c,
                                                 false, false);
}

DEVFN v8f vzero8() {
  v8f z;
#pragma unroll
  for (int i = 0; i < 8; ++i) z[i] = 0.0f;
  return z;
}

DEVFN float swishf(float x) {
  return x * __builtin_amdgcn_rcpf(1.0f + __expf(-x));
}

// ---------------------------------------------------------------------------
// Utility kernels
// ---------------------------------------------------------------------------

// Pack fp32 W[K,64] into bf16 WMMA-B fragment order.
__global__ void k_packB(const float* __restrict__ W, __bf16* __restrict__ P, int total) {
  int i = blockIdx.x * blockDim.x + threadIdx.x;
  if (i >= total) return;
  int e    = i & 15;
  int lane = (i >> 4) & 31;
  int nt   = (i >> 9) & 3;
  int kt   = i >> 11;
  int col  = nt * 16 + (lane & 15);
  int k    = kt * 32 + e + ((e >= 8) ? 8 : 0) + ((lane >> 4) ? 8 : 0);
  P[i] = (__bf16)W[k * 64 + col];
}

// Same, but fold a per-row (K) BN scale into the weights: P = (sc_k * W[k,n]).
__global__ void k_packBs(const float* __restrict__ W, const float* __restrict__ scsh,
                         __bf16* __restrict__ P, int total) {
  int i = blockIdx.x * blockDim.x + threadIdx.x;
  if (i >= total) return;
  int e    = i & 15;
  int lane = (i >> 4) & 31;
  int nt   = (i >> 9) & 3;
  int kt   = i >> 11;
  int col  = nt * 16 + (lane & 15);
  int k    = kt * 32 + e + ((e >= 8) ? 8 : 0) + ((lane >> 4) ? 8 : 0);
  P[i] = (__bf16)(W[k * 64 + col] * scsh[k]);
}

// Fold BN shift into the downstream bias: bout[n] = b[n] + sum_k sh_k * W[k,n].
__global__ void k_fold64(const float* __restrict__ W, const float* __restrict__ scsh,
                         const float* __restrict__ b, float* __restrict__ bout) {
  int n = threadIdx.x;
  if (n >= 64) return;
  float acc = b[n];
  for (int k = 0; k < 64; ++k) acc += scsh[64 + k] * W[k * 64 + n];
  bout[n] = acc;
}

// Fold BN into the [64,3] head: Wout = diag(sc) @ W ; bout = b + sh @ W.
__global__ void k_fold3(const float* __restrict__ W, const float* __restrict__ scsh,
                        const float* __restrict__ b, float* __restrict__ Wout,
                        float* __restrict__ bout) {
  int i = threadIdx.x;
  if (i < 192) Wout[i] = W[i] * scsh[i / 3];
  if (i < 3) {
    float acc = b[i];
    for (int k = 0; k < 64; ++k) acc += scsh[64 + k] * W[k * 3 + i];
    bout[i] = acc;
  }
}

__global__ void k_count(const int* __restrict__ dstI, float* __restrict__ cnt, int Etot) {
  int e = blockIdx.x * blockDim.x + threadIdx.x;
  if (e < Etot) atomicAdd(&cnt[dstI[e]], 1.0f);
}

// stats: [0..C) = sum, [C..2C) = sumsq.  scsh: [0..C) = scale, [C..2C) = shift.
__global__ void k_bn_fin(const float* __restrict__ stats, const float* __restrict__ g,
                         const float* __restrict__ beta, float count,
                         float* __restrict__ scsh, int C) {
  int i = threadIdx.x;
  if (i >= C) return;
  float mean = stats[i] / count;
  float var  = stats[C + i] / count - mean * mean;
  float inv  = rsqrtf(var + 1e-5f);
  float sc   = g[i] * inv;
  scsh[i]     = sc;
  scsh[C + i] = beta[i] - mean * sc;
}

__global__ void k_bn_apply(const __bf16* __restrict__ hpre, const float* __restrict__ scsh,
                           float* __restrict__ f32out, __bf16* __restrict__ bfout, int n) {
  int i = blockIdx.x * blockDim.x + threadIdx.x;
  if (i >= n) return;
  int c = i & 63;
  float v = (float)hpre[i] * scsh[c] + scsh[64 + c];
  if (f32out) f32out[i] = v;
  bfout[i] = (__bf16)v;
}

__global__ void k_bn_apply_out(float* __restrict__ out, const float* __restrict__ scsh2, int n) {
  int i = blockIdx.x * blockDim.x + threadIdx.x;
  if (i >= n) return;
  int c = i % 3;
  out[i] = out[i] * scsh2[c] + scsh2[3 + c];
}

// ---------------------------------------------------------------------------
// Encoder: h = swish(in[M,32] @ W[32,64] + b); store h (bf16) + BN stats.
// ---------------------------------------------------------------------------
__global__ __launch_bounds__(256) void k_encoder(
    const float* __restrict__ in, const __bf16* __restrict__ Wp,
    const float* __restrict__ bias, __bf16* __restrict__ hpre,
    float* __restrict__ stats, int M) {
  __shared__ float bsum[64], bsq[64];
  int tid = threadIdx.x;
  if (tid < 64) bsum[tid] = 0.0f;
  else if (tid < 128) bsq[tid - 64] = 0.0f;
  __syncthreads();

  int wave = tid >> 5, lane = tid & 31;
  int tile = blockIdx.x * 8 + wave;
  if (tile < (M >> 4)) {
    int r = lane & 15, hi = (lane >> 4) & 1;
    const float* rowp = in + (tile * 16 + r) * 32;
    v16bf a;
#pragma unroll
    for (int e = 0; e < 16; ++e) a[e] = (__bf16)rowp[a_k_of(e, hi)];
#pragma unroll
    for (int nt = 0; nt < 4; ++nt) {
      v8f acc = vzero8();
      acc = wmma_bf16(a, load_bp(Wp, 0, nt, lane), acc);
      int n = nt * 16 + r;
      float bn = bias[n];
      float s1 = 0.0f, s2 = 0.0f;
#pragma unroll
      for (int rr = 0; rr < 8; ++rr) {
        float h = swishf(acc[rr] + bn);
        hpre[(tile * 16 + rr + 8 * hi) * 64 + n] = (__bf16)h;
        s1 += h; s2 += h * h;
      }
      atomicAdd(&bsum[n], s1);
      atomicAdd(&bsq[n], s2);
    }
  }
  __syncthreads();
  if (tid < 64) atomicAdd(&stats[tid], bsum[tid]);
  else if (tid < 128) atomicAdd(&stats[tid], bsq[tid - 64]);
}

// ---------------------------------------------------------------------------
// Message GEMM1: A = [ne[dst] | ne[src]]  ([E,128]) @ W1[128,64] + b, swish.
// ---------------------------------------------------------------------------
__global__ __launch_bounds__(256) void k_msg1(
    const __bf16* __restrict__ ne, const int* __restrict__ srcI,
    const int* __restrict__ dstI, const __bf16* __restrict__ Wp,
    const float* __restrict__ bias, __bf16* __restrict__ hpre,
    float* __restrict__ stats, int Etot) {
  __shared__ __bf16 atile[8][16][128];  // 32 KB
  __shared__ float bsum[64], bsq[64];
  int tid = threadIdx.x;
  if (tid < 64) bsum[tid] = 0.0f;
  else if (tid < 128) bsq[tid - 64] = 0.0f;

  int wave = tid >> 5, lane = tid & 31;
  int tile = blockIdx.x * 8 + wave;
  bool act = tile < (Etot >> 4);
  if (act) {
    int r = lane & 15;
    int e = tile * 16 + r;
    int node = (lane >= 16) ? srcI[e] : dstI[e];  // cols 0-63: ne[dst]; 64-127: ne[src]
    const uint4* p = (const uint4*)(ne + node * 64);
    uint4* q = (uint4*)&atile[wave][r][(lane >= 16) ? 64 : 0];
#pragma unroll
    for (int i = 0; i < 8; ++i) q[i] = p[i];
  }
  __syncthreads();
  if (act) {
    int r = lane & 15, hi = (lane >> 4) & 1;
    v16bf afr[4];
#pragma unroll
    for (int kt = 0; kt < 4; ++kt)
      afr[kt] = load_a_tile(&atile[wave][0][0], 128, kt * 32, lane);
#pragma unroll
    for (int nt = 0; nt < 4; ++nt) {
      v8f acc = vzero8();
#pragma unroll
      for (int kt = 0; kt < 4; ++kt)
        acc = wmma_bf16(afr[kt], load_bp(Wp, kt, nt, lane), acc);
      int n = nt * 16 + r;
      float bn = bias[n];
      float s1 = 0.0f, s2 = 0.0f;
#pragma unroll
      for (int rr = 0; rr < 8; ++rr) {
        float h = swishf(acc[rr] + bn);
        hpre[(tile * 16 + rr + 8 * hi) * 64 + n] = (__bf16)h;
        s1 += h; s2 += h * h;
      }
      atomicAdd(&bsum[n], s1);
      atomicAdd(&bsq[n], s2);
    }
  }
  __syncthreads();
  if (tid < 64) atomicAdd(&stats[tid], bsum[tid]);
  else if (tid < 128) atomicAdd(&stats[tid], bsq[tid - 64]);
}

// ---------------------------------------------------------------------------
// Message GEMM2 + scatter: m = swish(h @ W2' + b2'); BN pre-folded into
// W2'/b2', so A is a plain bf16 load.  atomic-add into s[dst].
// ---------------------------------------------------------------------------
__global__ __launch_bounds__(256) void k_msg2(
    const __bf16* __restrict__ hpre, const __bf16* __restrict__ Wp,
    const float* __restrict__ bias, const int* __restrict__ dstI,
    float* __restrict__ sbuf, int Etot) {
  int tid = threadIdx.x, wave = tid >> 5, lane = tid & 31;
  int tile = blockIdx.x * 8 + wave;
  if (tile >= (Etot >> 4)) return;
  int r = lane & 15, hi = (lane >> 4) & 1;
  const __bf16* base = hpre + tile * 1024;  // 16 rows x 64
  v16bf a0 = load_a_tile(base, 64, 0, lane);
  v16bf a1 = load_a_tile(base, 64, 32, lane);
  int dsts[8];
#pragma unroll
  for (int rr = 0; rr < 8; ++rr) dsts[rr] = dstI[tile * 16 + rr + 8 * hi];
#pragma unroll
  for (int nt = 0; nt < 4; ++nt) {
    v8f acc = vzero8();
    acc = wmma_bf16(a0, load_bp(Wp, 0, nt, lane), acc);
    acc = wmma_bf16(a1, load_bp(Wp, 1, nt, lane), acc);
    int n = nt * 16 + r;
    float bn = bias[n];
#pragma unroll
    for (int rr = 0; rr < 8; ++rr) {
      float m = swishf(acc[rr] + bn);
      atomicAdd(&sbuf[dsts[rr] * 64 + n], m);
    }
  }
}

// ---------------------------------------------------------------------------
// Node update: u = swish(swish([ne | agg] @ W1 + b1) @ W2 + b2); ne += u.
// ---------------------------------------------------------------------------
__global__ __launch_bounds__(256) void k_upd(
    float* __restrict__ nef, __bf16* __restrict__ neb,
    const float* __restrict__ sbuf, const float* __restrict__ cnt,
    const __bf16* __restrict__ W1p, const float* __restrict__ b1,
    const __bf16* __restrict__ W2p, const float* __restrict__ b2, int Ntot) {
  __shared__ __bf16 utile[8][16][64];  // 16 KB
  int tid = threadIdx.x, wave = tid >> 5, lane = tid & 31;
  int tile = blockIdx.x * 8 + wave;
  bool act = tile < (Ntot >> 4);
  int r = lane & 15, hi = (lane >> 4) & 1;
  if (act) {
    int node = tile * 16 + r;
    float rc = __builtin_amdgcn_rcpf(fmaxf(cnt[node], 1.0f));
    v16bf afr[4];
#pragma unroll
    for (int kt = 0; kt < 4; ++kt) {
#pragma unroll
      for (int e = 0; e < 16; ++e) {
        int k = kt * 32 + a_k_of(e, hi);
        float v = (kt < 2) ? (float)neb[node * 64 + k]
                           : sbuf[node * 64 + (k - 64)] * rc;
        afr[kt][e] = (__bf16)v;
      }
    }
#pragma unroll
    for (int nt = 0; nt < 4; ++nt) {
      v8f acc = vzero8();
#pragma unroll
      for (int kt = 0; kt < 4; ++kt)
        acc = wmma_bf16(afr[kt], load_bp(W1p, kt, nt, lane), acc);
      int n = nt * 16 + r;
      float bn = b1[n];
#pragma unroll
      for (int rr = 0; rr < 8; ++rr)
        utile[wave][rr + 8 * hi][n] = (__bf16)swishf(acc[rr] + bn);
    }
  }
  __syncthreads();
  if (act) {
#pragma unroll
    for (int nt = 0; nt < 4; ++nt) {
      v8f acc = vzero8();
#pragma unroll
      for (int kt = 0; kt < 2; ++kt)
        acc = wmma_bf16(load_a_tile(&utile[wave][0][0], 64, kt * 32, lane),
                        load_bp(W2p, kt, nt, lane), acc);
      int n = nt * 16 + r;
      float bn = b2[n];
#pragma unroll
      for (int rr = 0; rr < 8; ++rr) {
        int idx = (tile * 16 + rr + 8 * hi) * 64 + n;
        float nv = nef[idx] + swishf(acc[rr] + bn);
        nef[idx] = nv;
        neb[idx] = (__bf16)nv;
      }
    }
  }
}

// ---------------------------------------------------------------------------
// Final GEMM1: A = [ne[dst]-ne[src] | ee] ([E,128]) @ agg0_W + b, swish.
// ---------------------------------------------------------------------------
__global__ __launch_bounds__(256) void k_final1(
    const float* __restrict__ nef, const __bf16* __restrict__ eeb,
    const int* __restrict__ srcI, const int* __restrict__ dstI,
    const __bf16* __restrict__ Wp, const float* __restrict__ bias,
    __bf16* __restrict__ hpre, float* __restrict__ stats, int Etot) {
  __shared__ __bf16 atile[8][16][128];
  __shared__ float bsum[64], bsq[64];
  int tid = threadIdx.x;
  if (tid < 64) bsum[tid] = 0.0f;
  else if (tid < 128) bsq[tid - 64] = 0.0f;

  int wave = tid >> 5, lane = tid & 31;
  int tile = blockIdx.x * 8 + wave;
  bool act = tile < (Etot >> 4);
  if (act) {
    int r = lane & 15;
    int e = tile * 16 + r;
    if (lane < 16) {
      int di = dstI[e] * 64, si = srcI[e] * 64;
#pragma unroll
      for (int c = 0; c < 64; ++c)
        atile[wave][r][c] = (__bf16)(nef[di + c] - nef[si + c]);
    } else {
      const uint4* p = (const uint4*)(eeb + e * 64);
      uint4* q = (uint4*)&atile[wave][r][64];
#pragma unroll
      for (int i = 0; i < 8; ++i) q[i] = p[i];
    }
  }
  __syncthreads();
  if (act) {
    int r = lane & 15, hi = (lane >> 4) & 1;
    v16bf afr[4];
#pragma unroll
    for (int kt = 0; kt < 4; ++kt)
      afr[kt] = load_a_tile(&atile[wave][0][0], 128, kt * 32, lane);
#pragma unroll
    for (int nt = 0; nt < 4; ++nt) {
      v8f acc = vzero8();
#pragma unroll
      for (int kt = 0; kt < 4; ++kt)
        acc = wmma_bf16(afr[kt], load_bp(Wp, kt, nt, lane), acc);
      int n = nt * 16 + r;
      float bn = bias[n];
      float s1 = 0.0f, s2 = 0.0f;
#pragma unroll
      for (int rr = 0; rr < 8; ++rr) {
        float h = swishf(acc[rr] + bn);
        hpre[(tile * 16 + rr + 8 * hi) * 64 + n] = (__bf16)h;
        s1 += h; s2 += h * h;
      }
      atomicAdd(&bsum[n], s1);
      atomicAdd(&bsq[n], s2);
    }
  }
  __syncthreads();
  if (tid < 64) atomicAdd(&stats[tid], bsum[tid]);
  else if (tid < 128) atomicAdd(&stats[tid], bsq[tid - 64]);
}

// ---------------------------------------------------------------------------
// Final GEMM2 (OUT=3): y = swish(g1 @ W3' + b3'), BN pre-folded into W3'/b3'.
// ---------------------------------------------------------------------------
__global__ __launch_bounds__(256) void k_final2(
    const __bf16* __restrict__ g1, const float* __restrict__ W,
    const float* __restrict__ b, float* __restrict__ out,
    float* __restrict__ stats2, int Etot) {
  __shared__ float ls[6];
  int tid = threadIdx.x;
  if (tid < 6) ls[tid] = 0.0f;
  __syncthreads();
  int e = blockIdx.x * blockDim.x + tid;
  if (e < Etot) {
    float y0 = b[0], y1 = b[1], y2 = b[2];
    const __bf16* row = g1 + e * 64;
#pragma unroll
    for (int k = 0; k < 64; ++k) {
      float v = (float)row[k];
      y0 += v * W[k * 3 + 0];
      y1 += v * W[k * 3 + 1];
      y2 += v * W[k * 3 + 2];
    }
    y0 = swishf(y0); y1 = swishf(y1); y2 = swishf(y2);
    out[e * 3 + 0] = y0;
    out[e * 3 + 1] = y1;
    out[e * 3 + 2] = y2;
    atomicAdd(&ls[0], y0); atomicAdd(&ls[1], y1); atomicAdd(&ls[2], y2);
    atomicAdd(&ls[3], y0 * y0); atomicAdd(&ls[4], y1 * y1); atomicAdd(&ls[5], y2 * y2);
  }
  __syncthreads();
  if (tid < 6) atomicAdd(&stats2[tid], ls[tid]);
}

// ---------------------------------------------------------------------------
// Host orchestration
// ---------------------------------------------------------------------------
extern "C" void kernel_launch(void* const* d_in, const int* in_sizes, int n_in,
                              void* d_out, int out_size, void* d_ws, size_t ws_size,
                              hipStream_t stream) {
  (void)in_sizes; (void)n_in; (void)out_size; (void)ws_size;

  const float* x       = (const float*)d_in[0];
  const float* ea      = (const float*)d_in[1];
  const int*   eidx    = (const int*)d_in[2];
  const int*   srcI    = eidx;
  const int*   dstI    = eidx + Ee;
  const float* node_W  = (const float*)d_in[3];
  const float* node_b  = (const float*)d_in[4];
  const float* node_g  = (const float*)d_in[5];
  const float* node_be = (const float*)d_in[6];
  const float* edge_W  = (const float*)d_in[7];
  const float* edge_b  = (const float*)d_in[8];
  const float* edge_g  = (const float*)d_in[9];
  const float* edge_be = (const float*)d_in[10];
  const float* msg_W1  = (const float*)d_in[11];
  const float* msg_b1  = (const float*)d_in[12];
  const float* msg_g   = (const float*)d_in[13];
  const float* msg_be  = (const float*)d_in[14];
  const float* msg_W2  = (const float*)d_in[15];
  const float* msg_b2  = (const float*)d_in[16];
  const float* upd_W1  = (const float*)d_in[17];
  const float* upd_b1  = (const float*)d_in[18];
  const float* upd_W2  = (const float*)d_in[19];
  const float* upd_b2  = (const float*)d_in[20];
  const float* agg0_W  = (const float*)d_in[21];
  const float* agg0_b  = (const float*)d_in[22];
  const float* agg0_g  = (const float*)d_in[23];
  const float* agg0_be = (const float*)d_in[24];
  const float* agg1_W  = (const float*)d_in[25];
  const float* agg1_b  = (const float*)d_in[26];
  const float* agg1_g  = (const float*)d_in[27];
  const float* agg1_be = (const float*)d_in[28];
  float* out = (float*)d_out;

  char* wp = (char*)d_ws;
  auto carve = [&](size_t b) -> void* {
    void* p = (void*)wp;
    wp += (b + 255) & ~(size_t)255;
    return p;
  };
  float*  ne_f   = (float*)carve((size_t)Nn * Hh * 4);
  __bf16* ne_b   = (__bf16*)carve((size_t)Nn * Hh * 2);
  __bf16* ee_b   = (__bf16*)carve((size_t)Ee * Hh * 2);
  __bf16* hpre   = (__bf16*)carve((size_t)Ee * Hh * 2);
  float*  sbuf   = (float*)carve((size_t)Nn * Hh * 4);
  float*  cntf   = (float*)carve((size_t)Nn * 4);
  float*  stats  = (float*)carve(128 * 4);
  float*  scsh   = (float*)carve(128 * 4);
  float*  stats2 = (float*)carve(64);
  float*  scsh2  = (float*)carve(64);
  __bf16* nWp    = (__bf16*)carve((size_t)32 * 64 * 2);
  __bf16* eWp    = (__bf16*)carve((size_t)32 * 64 * 2);
  __bf16* mW1p   = (__bf16*)carve((size_t)3 * 128 * 64 * 2);
  __bf16* uW1p   = (__bf16*)carve((size_t)3 * 128 * 64 * 2);
  __bf16* uW2p   = (__bf16*)carve((size_t)3 * 64 * 64 * 2);
  __bf16* a0Wp   = (__bf16*)carve((size_t)128 * 64 * 2);
  __bf16* w2s    = (__bf16*)carve((size_t)64 * 64 * 2);  // BN-folded msg_W2[l]
  float*  b2s    = (float*)carve(64 * 4);                // BN-folded msg_b2[l]
  float*  w3s    = (float*)carve(192 * 4);               // BN-folded agg1_W
  float*  b3s    = (float*)carve(16);                    // BN-folded agg1_b

  auto pack = [&](const float* src, __bf16* dst, int K) {
    int total = K * 64;
    k_packB<<<(total + 255) / 256, 256, 0, stream>>>(src, dst, total);
  };
  pack(node_W, nWp, 32);
  pack(edge_W, eWp, 32);
  for (int l = 0; l < 3; ++l) {
    pack(msg_W1 + l * 128 * 64, mW1p + l * 128 * 64, 128);
    pack(upd_W1 + l * 128 * 64, uW1p + l * 128 * 64, 128);
    pack(upd_W2 + l * 64 * 64,  uW2p + l * 64 * 64,  64);
  }
  pack(agg0_W, a0Wp, 128);

  hipMemsetAsync(cntf, 0, (size_t)Nn * 4, stream);
  k_count<<<(Ee + 255) / 256, 256, 0, stream>>>(dstI, cntf, Ee);

  const int nblkN = ((Nn / 16) + 7) / 8;  // 157
  const int nblkE = ((Ee / 16) + 7) / 8;  // 2500

  // Node encoder + BN
  hipMemsetAsync(stats, 0, 128 * 4, stream);
  k_encoder<<<nblkN, 256, 0, stream>>>(x, nWp, node_b, hpre, stats, Nn);
  k_bn_fin<<<1, 64, 0, stream>>>(stats, node_g, node_be, (float)Nn, scsh, 64);
  k_bn_apply<<<((Nn * 64) + 255) / 256, 256, 0, stream>>>(hpre, scsh, ne_f, ne_b, Nn * 64);

  // Edge encoder + BN
  hipMemsetAsync(stats, 0, 128 * 4, stream);
  k_encoder<<<nblkE, 256, 0, stream>>>(ea, eWp, edge_b, hpre, stats, Ee);
  k_bn_fin<<<1, 64, 0, stream>>>(stats, edge_g, edge_be, (float)Ee, scsh, 64);
  k_bn_apply<<<((Ee * 64) + 255) / 256, 256, 0, stream>>>(hpre, scsh, (float*)nullptr,
                                                          ee_b, Ee * 64);

  // Message-passing layers
  for (int l = 0; l < 3; ++l) {
    hipMemsetAsync(stats, 0, 128 * 4, stream);
    k_msg1<<<nblkE, 256, 0, stream>>>(ne_b, srcI, dstI, mW1p + l * 128 * 64,
                                      msg_b1 + l * 64, hpre, stats, Ee);
    k_bn_fin<<<1, 64, 0, stream>>>(stats, msg_g + l * 64, msg_be + l * 64, (float)Ee,
                                   scsh, 64);
    // Fold BN(h) into W2/b2 for this layer, then run GEMM2 + scatter.
    k_packBs<<<(64 * 64 + 255) / 256, 256, 0, stream>>>(msg_W2 + l * 64 * 64, scsh,
                                                        w2s, 64 * 64);
    k_fold64<<<1, 64, 0, stream>>>(msg_W2 + l * 64 * 64, scsh, msg_b2 + l * 64, b2s);
    hipMemsetAsync(sbuf, 0, (size_t)Nn * Hh * 4, stream);
    k_msg2<<<nblkE, 256, 0, stream>>>(hpre, w2s, b2s, dstI, sbuf, Ee);
    k_upd<<<nblkN, 256, 0, stream>>>(ne_f, ne_b, sbuf, cntf, uW1p + l * 128 * 64,
                                     upd_b1 + l * 64, uW2p + l * 64 * 64,
                                     upd_b2 + l * 64, Nn);
  }

  // Final readout
  hipMemsetAsync(stats, 0, 128 * 4, stream);
  k_final1<<<nblkE, 256, 0, stream>>>(ne_f, ee_b, srcI, dstI, a0Wp, agg0_b, hpre, stats, Ee);
  k_bn_fin<<<1, 64, 0, stream>>>(stats, agg0_g, agg0_be, (float)Ee, scsh, 64);
  k_fold3<<<1, 192, 0, stream>>>(agg1_W, scsh, agg1_b, w3s, b3s);

  hipMemsetAsync(stats2, 0, 6 * 4, stream);
  k_final2<<<(Ee + 255) / 256, 256, 0, stream>>>(hpre, w3s, b3s, out, stats2, Ee);
  k_bn_fin<<<1, 64, 0, stream>>>(stats2, agg1_g, agg1_be, (float)Ee, scsh2, 3);
  k_bn_apply_out<<<((Ee * 3) + 255) / 256, 256, 0, stream>>>(out, scsh2, Ee * 3);
}
